// SelfAttention_65352222376668
// MI455X (gfx1250) — compile-verified, matrix operand order checked by reading, not run on hardware
//
#include <hip/hip_runtime.h>
#include <hip/hip_bf16.h>

// ---------------------------------------------------------------------------
// Self-attention (B=4, S=4096, D=128), fp32 in/out, bf16 WMMA internally.
// Flash-attention structure; K/V staged via CDNA5 async-to-LDS double buffer.
// ---------------------------------------------------------------------------

typedef __attribute__((ext_vector_type(16))) __bf16 v16bf;
typedef __attribute__((ext_vector_type(8)))  __bf16 v8bf;
typedef __attribute__((ext_vector_type(8)))  float  v8f;

#define BATCH 4
#define SEQ   4096
#define DIM   128
#define NTOK  (BATCH * SEQ)   // 16384
#define PSTR  40              // LDS stride (bf16) for P tile rows  (80 B, 16B-mult)
#define KSTR  136             // LDS stride (bf16) for K tile rows  (272 B, 16B-mult, != 256B period)
#define VSTR  40              // LDS stride (bf16) for V tile rows  (80 B)
#define NCH   (SEQ / 32)      // 128 key chunks
#define KTILE (32 * KSTR)     // one K LDS tile (elems)
#define VTILE (128 * VSTR)    // one V LDS tile (elems)

// Load a 16x32 bf16 WMMA A/B fragment (ISA 7.12.2 layout).
// For A: row = M (lane&15). For B: row = N (lane&15).
// K runs: lanes 0-15 -> K = {k0+0..7, k0+16..23}; lanes 16-31 -> +8.
template <typename PT>
static __device__ __forceinline__ v16bf load_frag(PT base, int ld, int row, int k0, int lane) {
    const int kb = ((lane >> 4) & 1) * 8;
    PT p = base + (size_t)row * ld + k0 + kb;
    v8bf lo = *(const v8bf*)(p);
    v8bf hi = *(const v8bf*)(p + 16);
    v16bf f;
#pragma unroll
    for (int i = 0; i < 8; ++i) { f[i] = lo[i]; f[i + 8] = hi[i]; }
    return f;
}

static __device__ __forceinline__ v8f wmma_bf16(v16bf a, v16bf b, v8f c) {
    return __builtin_amdgcn_wmma_f32_16x16x32_bf16(false, a, false, b, (short)0, c, false, false);
}

// CDNA5 async copy: 16B global -> LDS, tracked by ASYNCcnt (per-lane address).
static __device__ __forceinline__ void async_ld_b128(const void* g, void* l) {
    const unsigned loff = (unsigned)(uintptr_t)l;   // generic LDS ptr: addr[31:0] = LDS offset
    asm volatile("global_load_async_to_lds_b128 %0, %1, off"
                 :: "v"(loff), "v"(g) : "memory");
}

// ---------------- prep: fp32 -> bf16 conversions ---------------------------

__global__ void cvt_x_kernel(const float* __restrict__ x, __bf16* __restrict__ xb, int n) {
    int i = blockIdx.x * blockDim.x + threadIdx.x;
    if (i < n) xb[i] = (__bf16)x[i];
}

// wt[m][o][i] = W_m[i][o]  (transposed so B fragments are contiguous in K=i)
__global__ void cvt_w_kernel(const float* __restrict__ Wq, const float* __restrict__ Wk,
                             const float* __restrict__ Wv, __bf16* __restrict__ wt) {
    int i = blockIdx.x * blockDim.x + threadIdx.x;   // 3*128*128 threads
    int m   = i >> 14;
    int rem = i & 16383;
    int o   = rem >> 7;
    int in  = rem & 127;
    const float* W = (m == 0) ? Wq : (m == 1) ? Wk : Wv;
    wt[i] = (__bf16)W[in * DIM + o];
}

// ---------------- QKV projection ------------------------------------------
// grid = NTOK/16 blocks, block = 128 threads (4 waves).
// Wave w handles N-tiles {2w, 2w+1} of q,k,v for the block's 16 tokens.
__global__ void __launch_bounds__(128)
proj_kernel(const __bf16* __restrict__ xb, const __bf16* __restrict__ wt,
            const float* __restrict__ bq, const float* __restrict__ bk,
            const float* __restrict__ bv,
            __bf16* __restrict__ qb, __bf16* __restrict__ kb, __bf16* __restrict__ vt) {
    const int lane = threadIdx.x & 31;
    const int wv   = threadIdx.x >> 5;
    const int lrow = lane & 15;
    const int hi   = lane >> 4;
    const int tok0 = blockIdx.x * 16;
    const int b    = tok0 >> 12;        // S = 4096
    const int s0   = tok0 & (SEQ - 1);

    v16bf a[4];
#pragma unroll
    for (int kc = 0; kc < 4; ++kc)
        a[kc] = load_frag(xb, DIM, tok0 + lrow, kc * 32, lane);

#pragma unroll
    for (int m = 0; m < 3; ++m) {
        const __bf16* w    = wt + m * DIM * DIM;
        const float*  bias = (m == 0) ? bq : (m == 1) ? bk : bv;
#pragma unroll
        for (int t = 0; t < 2; ++t) {
            const int nt  = wv * 2 + t;
            const int col = nt * 16 + lrow;
            const float bz = bias[col];
            v8f acc;
#pragma unroll
            for (int r = 0; r < 8; ++r) acc[r] = bz;
#pragma unroll
            for (int kc = 0; kc < 4; ++kc) {
                v16bf bf = load_frag(w, DIM, col, kc * 32, lane);
                acc = wmma_bf16(a[kc], bf, acc);
            }
            if (m < 2) {
                __bf16* dst = (m == 0) ? qb : kb;
#pragma unroll
                for (int r = 0; r < 8; ++r)
                    dst[(size_t)(tok0 + r + 8 * hi) * DIM + col] = (__bf16)acc[r];
            } else {
                // v transposed: vt[b][d][s]
                __bf16* dst = vt + (size_t)b * DIM * SEQ + (size_t)col * SEQ + s0;
#pragma unroll
                for (int r = 0; r < 8; ++r)
                    dst[r + 8 * hi] = (__bf16)acc[r];
            }
        }
    }
}

// ---------------- flash attention -----------------------------------------
// grid = NTOK/64 blocks, block = 128 threads (4 waves), one batch per block.
// All 4 waves share K/V tiles staged into LDS with async copies,
// double-buffered across 32-key chunks.
__global__ void __launch_bounds__(128)
attn_kernel(const __bf16* __restrict__ qb, const __bf16* __restrict__ kb,
            const __bf16* __restrict__ vt, const float* __restrict__ x,
            float* __restrict__ out) {
    // [ K tiles: 2 x KTILE ][ V tiles: 2 x VTILE ][ P tiles: 4 x 16 x PSTR ]
    __shared__ __bf16 smem[2 * KTILE + 2 * VTILE + 4 * 16 * PSTR];

    const int tid  = threadIdx.x;
    const int lane = tid & 31;
    const int wv   = tid >> 5;
    const int lrow = lane & 15;
    const int hi   = lane >> 4;
    const int tok0 = (blockIdx.x * 4 + wv) * 16;
    const int b    = (blockIdx.x * 64) >> 12;   // same batch for whole block

    const __bf16* kbb = kb + (size_t)b * SEQ * DIM;
    const __bf16* vtb = vt + (size_t)b * DIM * SEQ;
    __bf16* pw = smem + 2 * KTILE + 2 * VTILE + wv * 16 * PSTR;

    // q fragments for this wave's 16 rows (K = full 128 dims, 4 chunks)
    v16bf qf[4];
#pragma unroll
    for (int kc = 0; kc < 4; ++kc)
        qf[kc] = load_frag(qb, DIM, tok0 + lrow, kc * 32, lane);

    // online-softmax state (exp2 domain); rows handled: r + 8*hi
    float mi[8], li[8];
    v8f o[8];
#pragma unroll
    for (int r = 0; r < 8; ++r) { mi[r] = -1e30f; li[r] = 0.0f; }
#pragma unroll
    for (int dt = 0; dt < 8; ++dt)
#pragma unroll
        for (int r = 0; r < 8; ++r) o[dt][r] = 0.0f;

    // scores scale: (1/sqrt(128)) * log2(e), so exp2f() gives softmax exp
    const float c = 0.08838834764831845f * 1.4426950408889634f;

    // cooperative async stage of one 32-key chunk (8 x b128 per thread)
    auto stage = [&](int key0, int buf) {
        __bf16* kdst = smem + buf * KTILE;              // runtime GEP, no constant
        __bf16* vdst = smem + 2 * KTILE + buf * VTILE;  // addrspacecast tables
        // K tile: 32 rows x 128 elems = 512 x 16B; thread does 4
#pragma unroll
        for (int j = 0; j < 4; ++j) {
            const int i   = tid * 4 + j;          // 0..511
            const int row = i >> 4, ch = i & 15;
            async_ld_b128(kbb + (size_t)(key0 + row) * DIM + ch * 8,
                          kdst + row * KSTR + ch * 8);
        }
        // V tile (transposed): 128 rows x 32 elems = 512 x 16B; thread does 4
#pragma unroll
        for (int j = 0; j < 4; ++j) {
            const int i   = tid * 4 + j;
            const int row = i >> 2, ch = i & 3;
            async_ld_b128(vtb + (size_t)row * SEQ + key0 + ch * 8,
                          vdst + row * VSTR + ch * 8);
        }
    };

    stage(0, 0);
    int cur = 0;

    for (int ch = 0; ch < NCH; ++ch) {
        const int last = (ch + 1 == NCH);
        if (!last) stage((ch + 1) * 32, cur ^ 1);
        // own oldest 8 transfers (chunk ch) complete; newest 8 may be in flight
        if (!last) asm volatile("s_wait_asynccnt 0x8" ::: "memory");
        else       asm volatile("s_wait_asynccnt 0x0" ::: "memory");
        __syncthreads();                      // all waves' chunk-ch data in LDS

        const __bf16* kc_l = smem + cur * KTILE;
        const __bf16* vc_l = smem + 2 * KTILE + cur * VTILE;

        // ---- S = q @ k^T for 32 keys (two 16-wide N tiles), from LDS ----
        v8f sa = {}, sb = {};
#pragma unroll
        for (int kc = 0; kc < 4; ++kc) {
            v16bf k0f = load_frag(kc_l, KSTR, lrow,      kc * 32, lane);
            v16bf k1f = load_frag(kc_l, KSTR, 16 + lrow, kc * 32, lane);
            sa = wmma_bf16(qf[kc], k0f, sa);
            sb = wmma_bf16(qf[kc], k1f, sb);
        }

        // ---- online softmax (row stats via lane-shuffle over 16 cols) ----
        float a0[8], a1[8], mx[8];
#pragma unroll
        for (int r = 0; r < 8; ++r) {
            a0[r] = sa[r] * c;
            a1[r] = sb[r] * c;
            mx[r] = fmaxf(a0[r], a1[r]);
        }
#pragma unroll
        for (int off = 1; off < 16; off <<= 1)
#pragma unroll
            for (int r = 0; r < 8; ++r)
                mx[r] = fmaxf(mx[r], __shfl_xor(mx[r], off, 32));

        float mn[8], sc[8], p0[8], p1[8], rs[8];
#pragma unroll
        for (int r = 0; r < 8; ++r) {
            mn[r] = fmaxf(mi[r], mx[r]);
            sc[r] = exp2f(mi[r] - mn[r]);
            p0[r] = exp2f(a0[r] - mn[r]);
            p1[r] = exp2f(a1[r] - mn[r]);
            rs[r] = p0[r] + p1[r];
        }
#pragma unroll
        for (int off = 1; off < 16; off <<= 1)
#pragma unroll
            for (int r = 0; r < 8; ++r)
                rs[r] += __shfl_xor(rs[r], off, 32);
#pragma unroll
        for (int r = 0; r < 8; ++r) {
            li[r] = li[r] * sc[r] + rs[r];
            mi[r] = mn[r];
        }
#pragma unroll
        for (int dt = 0; dt < 8; ++dt)
#pragma unroll
            for (int r = 0; r < 8; ++r) o[dt][r] *= sc[r];

        // ---- re-layout P (D-frag f32) -> A-frag bf16 through LDS ----
#pragma unroll
        for (int r = 0; r < 8; ++r) {
            pw[(r + 8 * hi) * PSTR + lrow]      = (__bf16)p0[r];
            pw[(r + 8 * hi) * PSTR + 16 + lrow] = (__bf16)p1[r];
        }
        asm volatile("s_wait_dscnt 0x0" ::: "memory");
        v16bf pf = load_frag((const __bf16*)pw, PSTR, lrow, 0, lane);

        // ---- O += P @ V (V tile in LDS, contiguous in key dim) ----
#pragma unroll
        for (int dt = 0; dt < 8; ++dt) {
            v16bf vf = load_frag(vc_l, VSTR, dt * 16 + lrow, 0, lane);
            o[dt] = wmma_bf16(pf, vf, o[dt]);
        }

        __syncthreads();   // all waves done reading buf[cur] before restage
        cur ^= 1;
    }

    // ---- epilogue: normalize, residual add, store fp32 ----
    float inv[8];
#pragma unroll
    for (int r = 0; r < 8; ++r) inv[r] = 1.0f / li[r];
#pragma unroll
    for (int dt = 0; dt < 8; ++dt) {
#pragma unroll
        for (int r = 0; r < 8; ++r) {
            const int row = tok0 + r + 8 * hi;
            const int cg  = dt * 16 + lrow;
            out[(size_t)row * DIM + cg] = o[dt][r] * inv[r] + x[(size_t)row * DIM + cg];
        }
    }
}

// ---------------------------------------------------------------------------

extern "C" void kernel_launch(void* const* d_in, const int* in_sizes, int n_in,
                              void* d_out, int out_size, void* d_ws, size_t ws_size,
                              hipStream_t stream) {
    const float* x  = (const float*)d_in[0];
    const float* Wq = (const float*)d_in[1];
    const float* bq = (const float*)d_in[2];
    const float* Wk = (const float*)d_in[3];
    const float* bk = (const float*)d_in[4];
    const float* Wv = (const float*)d_in[5];
    const float* bv = (const float*)d_in[6];
    float* out = (float*)d_out;

    // workspace layout (bf16): xb | qb | kb | vt | wt(q,k,v transposed)
    __bf16* xb = (__bf16*)d_ws;
    __bf16* qb = xb + (size_t)NTOK * DIM;
    __bf16* kb = qb + (size_t)NTOK * DIM;
    __bf16* vt = kb + (size_t)NTOK * DIM;
    __bf16* wt = vt + (size_t)NTOK * DIM;   // 3 * 128 * 128

    const int nx = NTOK * DIM;              // 2,097,152
    cvt_x_kernel<<<(nx + 255) / 256, 256, 0, stream>>>(x, xb, nx);
    cvt_w_kernel<<<(3 * DIM * DIM) / 256, 256, 0, stream>>>(Wq, Wk, Wv, wt);

    proj_kernel<<<NTOK / 16, 128, 0, stream>>>(xb, wt, bq, bk, bv, qb, kb, vt);
    attn_kernel<<<NTOK / 64, 128, 0, stream>>>(qb, kb, vt, x, out);
}
